// AutoregressiveModel_19851338842563
// MI455X (gfx1250) — matrix operand order repaired and checked by reading, not compiled, not run on hardware
//
#include <hip/hip_runtime.h>
#include <hip/hip_bf16.h>

typedef __attribute__((ext_vector_type(16))) _Float16 v16h;
typedef __attribute__((ext_vector_type(8)))  float    v8f;

#define NB   64      // batch
#define CCH  3       // channels
#define HH   12
#define WW   12
#define KVAL 128     // categorical values
#define HW   144
#define NOUT 384     // K*C output channels
#define KPAD 66      // padded LDS row stride (>= 64, bank-stagger)
#define NTAP 36      // live masked taps: 2 full rows (10) + 2 of center row, x3 channels

union FragH { v16h v; unsigned u[8]; };

__device__ __forceinline__ unsigned pcg_hash(unsigned x) {
    x = x * 747796405u + 2891336453u;
    unsigned w = ((x >> ((x >> 28) + 4u)) ^ x) * 277803737u;
    return (w >> 22) ^ w;
}

__launch_bounds__(256, 1)
__global__ void pixelcnn_sample(const float* __restrict__ cond,
                                const float* __restrict__ Wc,
                                const float* __restrict__ bias,
                                float* __restrict__ out)
{
    // LDS-resident state (~163 KB of CDNA5's 320 KB/WGP)
    __shared__ _Float16 sImg[CCH * HW * NB];   // [c][p][n]  image, f16-exact ints
    __shared__ _Float16 sBt[NOUT * KPAD];      // [o][k]     masked weights, K-major
    __shared__ _Float16 sA[NB * KPAD];         // [m][k]     per-step patch matrix
    __shared__ _Float16 sD[NOUT * KPAD];       // [o][n]     per-step logits
    __shared__ float    sBias[NOUT];

    const int tid  = threadIdx.x;
    const int lane = tid & 31;
    const int wave = tid >> 5;
    const int lg   = lane >> 4;   // half-wave group (ISA A/B/D layouts)
    const int ll   = lane & 15;

    // ---------- one-time staging: image, masked f16 weights, bias ----------
    for (int e = tid; e < CCH * HW * NB; e += 256) {
        int c = e / (HW * NB);
        int r = e - c * HW * NB;
        int p = r / NB;
        int n = r - p * NB;
        sImg[e] = (_Float16)cond[(n * CCH + c) * HW + p];
    }
    for (int e = tid; e < NOUT * KPAD; e += 256) {
        int o  = e / KPAD;
        int kt = e - o * KPAD;
        float v = 0.f;
        if (kt < NTAP) {   // kt -> (ch, tap) -> (dy, dx) under the type-A mask
            int ch  = kt / 12;
            int tap = kt - ch * 12;
            int dy  = (tap < 10) ? (tap / 5) : 2;
            int dx  = (tap < 10) ? (tap % 5) : (tap - 10);
            v = Wc[((o * CCH + ch) * 5 + dy) * 5 + dx];
        }
        sBt[e] = (_Float16)v;
    }
    for (int e = tid; e < NOUT; e += 256) sBias[e] = bias[e];
    __syncthreads();

    // ---------- persistent B fragments: wave owns 3 N-tiles x 2 K-steps ----------
    // B 16-bit 32x16 layout: lane = column N (ll); VGPR 0..3 hold K pairs in [lg*8, lg*8+7], VGPR 4..7 hold [16+lg*8, 23+lg*8].
    FragH Bfrag[3][2];
#pragma unroll
    for (int j = 0; j < 3; ++j)
#pragma unroll
        for (int ks = 0; ks < 2; ++ks) {
            int o = (wave * 3 + j) * 16 + ll;
#pragma unroll
            for (int v = 0; v < 8; ++v) {
                int k = ((v < 4) ? (v * 2 + lg * 8) : (16 + (v - 4) * 2 + lg * 8)) + ks * 32;
                Bfrag[j][ks].u[v] = *(const unsigned*)&sBt[o * KPAD + k];
            }
        }

    // ---------- autoregressive raster loop (sequential by construction) ----------
    for (int i = 0; i < HW; ++i) {
        const int pr = i / WW;
        const int pc = i - pr * WW;

        // Build A[64 x 64]: row = batch, col = (ch,tap); each thread fills 16 halves.
        {
            int m  = tid >> 2;
            int k0 = (tid & 3) * 16;
#pragma unroll
            for (int jj = 0; jj < 16; ++jj) {
                int kt = k0 + jj;
                float v = 0.f;
                if (kt < NTAP) {
                    int ch  = kt / 12;
                    int tap = kt - ch * 12;
                    int dy  = (tap < 10) ? (tap / 5) : 2;
                    int dx  = (tap < 10) ? (tap % 5) : (tap - 10);
                    int y = pr + dy - 2;   // always < HH (dy <= 2)
                    int x = pc + dx - 2;
                    if (y >= 0 && x >= 0 && x < WW)
                        v = (float)sImg[(ch * HW + y * WW + x) * NB + m];
                }
                sA[m * KPAD + kt] = (_Float16)v;
            }
        }
        __syncthreads();

        // GEMM D[64x384] = A[64x64] * B[64x384]; 24 WMMAs per wave per step.
        v8f acc[4][3];
#pragma unroll
        for (int mt = 0; mt < 4; ++mt)
#pragma unroll
            for (int j = 0; j < 3; ++j)
                acc[mt][j] = (v8f){0.f, 0.f, 0.f, 0.f, 0.f, 0.f, 0.f, 0.f};

#pragma unroll
        for (int ks = 0; ks < 2; ++ks) {
            // A 16-bit 16x32 layout: lane = row M (ll); VGPR 0..3 hold K pairs in
            // [lg*8, lg*8+7], VGPR 4..7 hold [16+lg*8, 23+lg*8].
            FragH Af[4];
#pragma unroll
            for (int mt = 0; mt < 4; ++mt) {
                int m = mt * 16 + ll;
#pragma unroll
                for (int v = 0; v < 8; ++v) {
                    int k = ((v < 4) ? (v * 2 + lg * 8)
                                     : (16 + (v - 4) * 2 + lg * 8)) + ks * 32;
                    Af[mt].u[v] = *(const unsigned*)&sA[m * KPAD + k];
                }
            }
#pragma unroll
            for (int mt = 0; mt < 4; ++mt)
#pragma unroll
                for (int j = 0; j < 3; ++j)
                    acc[mt][j] = __builtin_amdgcn_wmma_f32_16x16x32_f16(
                        false, Af[mt].v, false, Bfrag[j][ks].v,
                        (short)0, acc[mt][j], false, false);
        }

        // D layout: VGPR q -> M = mt*16 + lg*8 + q, lane -> N(o). Spill to LDS f16.
#pragma unroll
        for (int j = 0; j < 3; ++j) {
            int o = (wave * 3 + j) * 16 + ll;
#pragma unroll
            for (int mt = 0; mt < 4; ++mt) {
                int m0 = mt * 16 + lg * 8;
#pragma unroll
                for (int q = 0; q < 8; ++q)
                    sD[o * KPAD + m0 + q] = (_Float16)acc[mt][j][q];
            }
        }
        __syncthreads();

        // Gumbel-max categorical sample: one thread per (batch n, channel c).
        if (tid < CCH * NB) {
            int n = tid & (NB - 1);
            int c = tid >> 6;
            unsigned seed = pcg_hash(0x9E3779B9u ^ (unsigned)(i * 1315423911)
                                                 ^ ((unsigned)tid * 2654435761u));
            float best = -1e30f;
            int   bi   = 0;
            for (int k = 0; k < KVAL; ++k) {
                int o = k * CCH + c;
                float logit = (float)sD[o * KPAD + n] + sBias[o];
                unsigned h = pcg_hash(seed + (unsigned)k);
                float u = ((float)(h >> 9) + 0.5f) * (1.0f / 8388608.0f); // (0,1)
                float val = logit - __logf(-__logf(u));
                if (val > best) { best = val; bi = k; }
            }
            float cur = (float)sImg[(c * HW + i) * NB + n];
            if (cur < 0.f) sImg[(c * HW + i) * NB + n] = (_Float16)bi;  // fill only unobserved
        }
        __syncthreads();
    }

    // ---------- write final image [n][c][p] as float ----------
    for (int e = tid; e < NB * CCH * HW; e += 256) {
        int n = e / (CCH * HW);
        int r = e - n * CCH * HW;
        int c = r / HW;
        int p = r - c * HW;
        out[e] = (float)sImg[(c * HW + p) * NB + n];
    }
}

extern "C" void kernel_launch(void* const* d_in, const int* in_sizes, int n_in,
                              void* d_out, int out_size, void* d_ws, size_t ws_size,
                              hipStream_t stream) {
    const float* cond = (const float*)d_in[0];   // conditioned_on [64,3,12,12]
    const float* Wc   = (const float*)d_in[1];   // [384,3,5,5]
    const float* b    = (const float*)d_in[2];   // [384]
    float* out = (float*)d_out;                  // [64,3,12,12]
    (void)in_sizes; (void)n_in; (void)out_size; (void)d_ws; (void)ws_size;
    // Sequential 144-step chain; all state LDS-resident -> single workgroup.
    pixelcnn_sample<<<1, 256, 0, stream>>>(cond, Wc, b, out);
}
